// VQ_VAE_z16_7859790152057
// MI455X (gfx1250) — matrix operand order repaired and checked by reading, not compiled, not run on hardware
//
#include <hip/hip_runtime.h>
#include <math.h>

typedef __attribute__((ext_vector_type(16))) _Float16 v16h;
typedef __attribute__((ext_vector_type(8)))  float    v8f;

#define WAVES_PER_BLOCK 8

// ---------------------------------------------------------------------------
// Generic implicit-GEMM conv on v_wmma_f32_16x16x32_f16.
//   M = N*OH*OW pixels (16/tile), N = Cout (16/tile), K = Cin*T (32/chunk)
//   Input transform (prev-layer BN scale/shift + ReLU) fused into A gather.
//   ConvT handled as per-parity forward conv (oh_mul/oh_add remap output).
//   T (taps per input channel) and f32-vs-f16 input are compile-time.
// ---------------------------------------------------------------------------
struct ConvP {
  const _Float16* __restrict__ in;    // f16 input activations
  const float*    __restrict__ in32;  // f32 input activations
  const _Float16* __restrict__ wpack; // packed B fragments [ntiles][chunks][32][16]
  const float*    __restrict__ bias;  // [Cout]
  const float*    __restrict__ ss;    // in-chan scale at [ci], shift at [32+ci] (nullable)
  _Float16*       __restrict__ out;   // f16 output (nullable)
  float*          __restrict__ out32; // f32 output (nullable, d3 -> d_out)
  const float*    __restrict__ lref;  // recon-loss reference (nullable)
  float*          __restrict__ lacc;  // recon-loss accumulator
  int N, Cin, IH, IW, Cout;
  int OH, OW, OHF, OWF;               // logical out dims; full out dims (parity remap)
  int stride, K, chunks, ntiles;
  int owShift, ohwShift;              // OW, OH*OW are powers of two
  int relu_in, relu_out;
  int oh_mul, oh_add, ow_mul, ow_add;
  unsigned long long dyP, dxP;        // 16 x 4-bit (offset+2) tap displacements
};

template <int T, bool IN32>
__global__ __launch_bounds__(256) void conv_wmma(ConvP p) {
  const int lane = threadIdx.x & 31;
  const int wv   = threadIdx.x >> 5;
  const int tiles = (p.N << p.ohwShift) >> 4;
  const int tile  = blockIdx.x * WAVES_PER_BLOCK + wv;
  __shared__ float lsum;
  if (p.lref) { if (threadIdx.x == 0) lsum = 0.f; __syncthreads(); }
  float myloss = 0.f;
  const int hl  = lane & 15;
  const int kh2 = lane >> 4;
  if (tile < tiles) {
    const int ohw  = 1 << p.ohwShift;
    const int owm  = (1 << p.owShift) - 1;
    const int m    = (tile << 4) + hl;
    const int n0   = m >> p.ohwShift;
    const int r0   = m & (ohw - 1);
    const int oh0  = r0 >> p.owShift;
    const int ow0  = r0 & owm;
    const int ihb  = oh0 * p.stride;
    const int iwb  = ow0 * p.stride;
    const size_t nbase = (size_t)n0 * p.Cin * p.IH * p.IW;
    for (int nt = 0; nt < p.ntiles; ++nt) {
      v8f acc = {0.f, 0.f, 0.f, 0.f, 0.f, 0.f, 0.f, 0.f};
      const _Float16* wp = p.wpack + (size_t)nt * p.chunks * 512;
      for (int c = 0; c < p.chunks; ++c) {
        // prefetch next chunk's B fragment line (global_prefetch_b8)
        if (c + 1 < p.chunks)
          __builtin_prefetch(wp + ((size_t)((c + 1) * 32 + lane) << 4), 0, 3);
        v16h a;
#pragma unroll
        for (int j = 0; j < 16; ++j) {
          // A k order used here (contiguous, not ISA table order): lanes 0-15 hold K=0..15;
          // lanes 16-31 hold K=16..31 (matches packed B).
          const int kl = kh2 * 16 + j;
          const int k  = c * 32 + kl;
          float v = 0.f;
          if (k < p.K) {
            const int ci = k / T;               // T compile-time -> const magic/shift
            const int t  = k - ci * T;
            const int ih = ihb + (int)((p.dyP >> (4 * t)) & 15ull) - 2;
            const int iw = iwb + (int)((p.dxP >> (4 * t)) & 15ull) - 2;
            if ((unsigned)ih < (unsigned)p.IH && (unsigned)iw < (unsigned)p.IW) {
              const size_t idx = nbase + ((size_t)ci * p.IH + ih) * p.IW + iw;
              v = IN32 ? p.in32[idx] : (float)p.in[idx];
              if (p.ss) v = fmaf(v, p.ss[ci], p.ss[32 + ci]);
              if (p.relu_in) v = fmaxf(v, 0.f);
            }
          }
          a[j] = (_Float16)v;
        }
        // B fragment: pre-packed so each lane reads 16 contiguous f16 (32B)
        const v16h b = *(const v16h*)(wp + ((size_t)(c * 32 + lane) << 4));
        acc = __builtin_amdgcn_wmma_f32_16x16x32_f16(false, a, false, b,
                                                     (short)0, acc, false, false);
      }
      // D layout: lane column = lane&15, vgpr i -> row (lane>>4)*8 + i
      const int co  = (nt << 4) + hl;
      const float bb = (co < p.Cout) ? p.bias[co] : 0.f;
#pragma unroll
      for (int i = 0; i < 8; ++i) {
        const int pr  = (tile << 4) + kh2 * 8 + i;
        const int n2  = pr >> p.ohwShift;
        const int r2  = pr & (ohw - 1);
        const int oh2 = r2 >> p.owShift;
        const int ow2 = r2 & owm;
        float v = acc[i] + bb;
        if (p.relu_out) v = fmaxf(v, 0.f);
        if (co < p.Cout) {
          const int ohF = oh2 * p.oh_mul + p.oh_add;
          const int owF = ow2 * p.ow_mul + p.ow_add;
          const size_t oidx = (((size_t)n2 * p.Cout + co) * p.OHF + ohF) * p.OWF + owF;
          if (p.out)   p.out[oidx]   = (_Float16)v;
          if (p.out32) p.out32[oidx] = v;
          if (p.lref)  { const float d = v - p.lref[oidx]; myloss = fmaf(d, d, myloss); }
        }
      }
    }
  }
  if (p.lref) {
    atomicAdd(&lsum, myloss);
    __syncthreads();
    if (threadIdx.x == 0) atomicAdd(p.lacc, lsum);
  }
}

// ---------------------------------------------------------------------------
// Weight packer: produce per-lane B fragments.
// B k order used here (contiguous, matching A gather): lane (n=lane&15, khalf=lane>>4), elem j = B[khalf*16+j][n]
// ---------------------------------------------------------------------------
struct PackP {
  const float* w;
  _Float16*    dst;
  int Cin, Cout, T, K, chunks, ntiles;
  int mode;          // 0: forward OIHW; 1: convT [I][O][4][4] parity taps
  int KH, KW;
  int tapH[2], tapW[2];
};

__global__ void pack_w(PackP p) {
  const int total = p.ntiles * p.chunks * 512;
  for (int s = blockIdx.x * blockDim.x + threadIdx.x; s < total;
       s += blockDim.x * gridDim.x) {
    const int j    = s & 15;
    const int lane = (s >> 4) & 31;
    const int cnt  = s >> 9;                 // nt*chunks + c
    const int c    = cnt % p.chunks;
    const int nt   = cnt / p.chunks;
    const int n    = nt * 16 + (lane & 15);
    const int k    = c * 32 + (lane >> 4) * 16 + j;
    float v = 0.f;
    if (k < p.K && n < p.Cout) {
      const int ci = k / p.T;
      const int t  = k - ci * p.T;
      if (p.mode == 0) {
        const int fh = t / p.KW, fw = t - (t / p.KW) * p.KW;
        v = p.w[(((size_t)n * p.Cin + ci) * p.KH + fh) * p.KW + fw];
      } else {
        const int th = t >> 1, tw = t & 1;
        v = p.w[(((size_t)ci * p.Cout + n) * 4 + p.tapH[th]) * 4 + p.tapW[tw]];
      }
    }
    p.dst[s] = (_Float16)v;
  }
}

// ---------------------------------------------------------------------------
// BN statistics: one block per (n,c) row, reduce HW, atomic into sums[c], sums[32+c]
// ---------------------------------------------------------------------------
__global__ __launch_bounds__(256) void bn_stats(const _Float16* act, float* sums,
                                                int cMask, int HW) {
  const int row = blockIdx.x;
  const int c   = row & cMask;
  const _Float16* pr = act + (size_t)row * HW;
  float s = 0.f, q = 0.f;
  for (int i = threadIdx.x; i < HW; i += blockDim.x) {
    const float v = (float)pr[i];
    s += v; q = fmaf(v, v, q);
  }
  __shared__ float sh[512];
  sh[threadIdx.x] = s; sh[256 + threadIdx.x] = q;
  __syncthreads();
  for (int st = 128; st > 0; st >>= 1) {
    if (threadIdx.x < st) {
      sh[threadIdx.x] += sh[threadIdx.x + st];
      sh[256 + threadIdx.x] += sh[256 + threadIdx.x + st];
    }
    __syncthreads();
  }
  if (threadIdx.x == 0) {
    atomicAdd(&sums[c], sh[0]);
    atomicAdd(&sums[32 + c], sh[256]);
  }
}

__global__ void bn_finalize(const float* sums, float* ss, const float* gamma,
                            const float* beta, int C, float invCnt) {
  const int c = threadIdx.x;
  if (c < C) {
    const float mean = sums[c] * invCnt;
    const float var  = sums[32 + c] * invCnt - mean * mean;
    const float sc   = gamma[c] * rsqrtf(var + 1e-5f);
    ss[c]      = sc;
    ss[32 + c] = beta[c] - mean * sc;
  }
}

__global__ void bn_apply_k(const _Float16* in, const float* ss, float* out,
                           int add, int hwShift, int cMask, int total) {
  for (int i = blockIdx.x * blockDim.x + threadIdx.x; i < total;
       i += blockDim.x * gridDim.x) {
    const int c = (i >> hwShift) & cMask;
    const float v = fmaf((float)in[i], ss[c], ss[32 + c]);
    out[i] = add ? out[i] + v : v;
  }
}

__global__ void zero_f32(float* p, int n) {
  const int i = blockIdx.x * blockDim.x + threadIdx.x;
  if (i < n) p[i] = 0.f;
}

// ---------------------------------------------------------------------------
// Vector quantizer: zq = nearest codeword; sum of min-dist = sum((q-z)^2);
// histogram of codes. z layout [n][16][256] f32, HW=256, K=64 codes.
// ---------------------------------------------------------------------------
__global__ __launch_bounds__(256) void vq_kernel(const float* z, const float* emb,
                                                 _Float16* zq, float* acc) {
  __shared__ float se[1024];
  __shared__ float lh[64];
  __shared__ float red[256];
  for (int i = threadIdx.x; i < 1024; i += 256) se[i] = emb[i];
  if (threadIdx.x < 64) lh[threadIdx.x] = 0.f;
  __syncthreads();
  const int p  = blockIdx.x * 256 + threadIdx.x;   // 131072 pixels exactly
  const int n  = p >> 8;
  const int hw = p & 255;
  const float* zp = z + ((size_t)n << 12) + hw;
  float x[16];
#pragma unroll
  for (int c = 0; c < 16; ++c) x[c] = zp[(size_t)c << 8];
  int best = 0; float bd = 3.4e38f;
  for (int k = 0; k < 64; ++k) {
    float d = 0.f;
#pragma unroll
    for (int c = 0; c < 16; ++c) { const float t = x[c] - se[(k << 4) + c]; d = fmaf(t, t, d); }
    if (d < bd) { bd = d; best = k; }
  }
  _Float16* qp = zq + ((size_t)n << 12) + hw;
#pragma unroll
  for (int c = 0; c < 16; ++c) qp[(size_t)c << 8] = (_Float16)se[(best << 4) + c];
  atomicAdd(&lh[best], 1.f);
  red[threadIdx.x] = bd;
  __syncthreads();
  for (int st = 128; st > 0; st >>= 1) {
    if (threadIdx.x < st) red[threadIdx.x] += red[threadIdx.x + st];
    __syncthreads();
  }
  if (threadIdx.x == 0) atomicAdd(&acc[0], red[0]);
  if (threadIdx.x < 64 && lh[threadIdx.x] != 0.f) atomicAdd(&acc[2 + threadIdx.x], lh[threadIdx.x]);
}

__global__ void final_losses(const float* acc, float* out_tail) {
  const float e_mean   = acc[0] / 2097152.0f;       // mean((q-z)^2)
  const float rec_mean = acc[1] / 16777216.0f;
  float s = 0.f;
  for (int k = 0; k < 64; ++k) {
    const float pk = acc[2 + k] / 131072.0f;
    s += pk * logf(pk + 1e-10f);
  }
  out_tail[0] = rec_mean + 1.25f * e_mean;          // recon + q_lat + 0.25*e_lat
  out_tail[1] = expf(-s);
}

// ---------------------------------------------------------------------------
// Host orchestration
// ---------------------------------------------------------------------------
static inline void dispatch_conv(const ConvP& p, int T, bool in32, int blocks,
                                 hipStream_t stream) {
  const dim3 g(blocks), b(256);
  if (in32) {
    if (T == 1)       conv_wmma<1,  true ><<<g, b, 0, stream>>>(p);
    else              conv_wmma<9,  true ><<<g, b, 0, stream>>>(p);
  } else {
    if (T == 1)       conv_wmma<1,  false><<<g, b, 0, stream>>>(p);
    else if (T == 4)  conv_wmma<4,  false><<<g, b, 0, stream>>>(p);
    else if (T == 9)  conv_wmma<9,  false><<<g, b, 0, stream>>>(p);
    else              conv_wmma<16, false><<<g, b, 0, stream>>>(p);
  }
}

extern "C" void kernel_launch(void* const* d_in, const int* in_sizes, int n_in,
                              void* d_out, int out_size, void* d_ws, size_t ws_size,
                              hipStream_t stream) {
  const float* inputs = (const float*)d_in[0];
  const float* c0_w = (const float*)d_in[1];  const float* c0_b = (const float*)d_in[2];
  const float* c1_w = (const float*)d_in[3];  const float* c1_b = (const float*)d_in[4];
  const float* bn1_g = (const float*)d_in[5]; const float* bn1_b = (const float*)d_in[6];
  const float* c2_w = (const float*)d_in[7];  const float* c2_b = (const float*)d_in[8];
  const float* bn2_g = (const float*)d_in[9]; const float* bn2_b = (const float*)d_in[10];
  const float* c3_w = (const float*)d_in[11]; const float* c3_b = (const float*)d_in[12];
  const float* bn3_g = (const float*)d_in[13]; const float* bn3_b = (const float*)d_in[14];
  const float* c4_w = (const float*)d_in[15]; const float* c4_b = (const float*)d_in[16];
  const float* bn4_g = (const float*)d_in[17]; const float* bn4_b = (const float*)d_in[18];
  const float* r0a_w = (const float*)d_in[19]; const float* r0a_b = (const float*)d_in[20];
  const float* r0g1 = (const float*)d_in[21];  const float* r0b1 = (const float*)d_in[22];
  const float* r0b_w = (const float*)d_in[23]; const float* r0b_b = (const float*)d_in[24];
  const float* r0g2 = (const float*)d_in[25];  const float* r0b2 = (const float*)d_in[26];
  const float* r1a_w = (const float*)d_in[27]; const float* r1a_b = (const float*)d_in[28];
  const float* r1g1 = (const float*)d_in[29];  const float* r1b1 = (const float*)d_in[30];
  const float* r1b_w = (const float*)d_in[31]; const float* r1b_b = (const float*)d_in[32];
  const float* r1g2 = (const float*)d_in[33];  const float* r1b2 = (const float*)d_in[34];
  const float* emb  = (const float*)d_in[35];
  const float* d0_w = (const float*)d_in[36]; const float* d0_b = (const float*)d_in[37];
  const float* d1_w = (const float*)d_in[38]; const float* d1_b = (const float*)d_in[39];
  const float* d2_w = (const float*)d_in[40]; const float* d2_b = (const float*)d_in[41];
  const float* d3_w = (const float*)d_in[42]; const float* d3_b = (const float*)d_in[43];
  float* dout = (float*)d_out;

  char* ws = (char*)d_ws;
  // workspace layout (aliased over time)
  constexpr size_t OFS_Z0   = 0;                        // z0 f16 128MiB; later y_d2
  constexpr size_t OFS_Y1   = OFS_Z0  + 134217728;      // y1 f16 32MiB; later y_d1
  constexpr size_t OFS_Y2   = OFS_Y1  + 33554432;       // y2 f16 16MiB; later y_d0
  constexpr size_t OFS_Y3   = OFS_Y2  + 16777216;       // y3 f16 4MiB
  constexpr size_t OFS_Y4   = OFS_Y3  + 4194304;        // y4 f16 4MiB
  constexpr size_t OFS_Z    = OFS_Y4  + 4194304;        // z   f32 8MiB
  constexpr size_t OFS_H32  = OFS_Z   + 8388608;        // h32 f16 8MiB
  constexpr size_t OFS_H16  = OFS_H32 + 8388608;        // h16 f16 4MiB
  constexpr size_t OFS_ZQ   = OFS_H16 + 4194304;        // zq  f16 4MiB
  constexpr size_t OFS_STAT = OFS_ZQ  + 4194304;        // stats/losses 8KB
  constexpr size_t OFS_PACK = OFS_STAT + 8192;          // 22 * 16KB weight packs

  _Float16* z0  = (_Float16*)(ws + OFS_Z0);
  _Float16* y1  = (_Float16*)(ws + OFS_Y1);
  _Float16* y2  = (_Float16*)(ws + OFS_Y2);
  _Float16* y3  = (_Float16*)(ws + OFS_Y3);
  _Float16* y4  = (_Float16*)(ws + OFS_Y4);
  float*    zf  = (float*)   (ws + OFS_Z);
  _Float16* h32 = (_Float16*)(ws + OFS_H32);
  _Float16* h16 = (_Float16*)(ws + OFS_H16);
  _Float16* zq  = (_Float16*)(ws + OFS_ZQ);
  _Float16* yd0 = (_Float16*)(ws + OFS_Y2);             // alias y2
  _Float16* yd1 = (_Float16*)(ws + OFS_Y1);             // alias y1
  _Float16* yd2 = (_Float16*)(ws + OFS_Z0);             // alias z0
  float* statf  = (float*)(ws + OFS_STAT);
  _Float16* packs = (_Float16*)(ws + OFS_PACK);

  auto sums = [&](int b) { return statf + b * 64; };
  auto ssb  = [&](int b) { return statf + 512 + b * 64; };
  float* acc = statf + 1024;                            // [0]=vq, [1]=recon, [2..65]=hist
  auto pk = [&](int slot) { return packs + (size_t)slot * 8192; };
  auto ilog2 = [](int x) { int s = 0; while ((1 << s) < x) ++s; return s; };

  zero_f32<<<dim3(8), 256, 0, stream>>>(statf, 2048);

  // ---- weight packing ----
  auto packFwd = [&](int slot, const float* w, int Cin, int Cout, int KH, int KW) {
    PackP q{}; q.w = w; q.dst = pk(slot); q.Cin = Cin; q.Cout = Cout;
    q.T = KH * KW; q.K = Cin * q.T; q.chunks = (q.K + 31) / 32;
    q.ntiles = (Cout + 15) / 16; q.mode = 0; q.KH = KH; q.KW = KW;
    pack_w<<<dim3(4), 256, 0, stream>>>(q);
  };
  auto packT = [&](int slot, const float* w, int Cin, int Cout, int ph, int pw) {
    PackP q{}; q.w = w; q.dst = pk(slot); q.Cin = Cin; q.Cout = Cout;
    q.T = 4; q.K = Cin * 4; q.chunks = (q.K + 31) / 32; q.ntiles = 1; q.mode = 1;
    q.tapH[0] = ph ? 2 : 1; q.tapH[1] = ph ? 0 : 3;
    q.tapW[0] = pw ? 2 : 1; q.tapW[1] = pw ? 0 : 3;
    pack_w<<<dim3(4), 256, 0, stream>>>(q);
  };
  packFwd(0, c0_w, 2, 8, 1, 1);
  packFwd(1, c1_w, 8, 8, 4, 4);
  packFwd(2, c2_w, 8, 16, 4, 4);
  packFwd(3, c3_w, 16, 16, 4, 4);
  packFwd(4, c4_w, 16, 16, 3, 3);
  packFwd(5, r0a_w, 16, 32, 3, 3);
  packFwd(6, r0b_w, 32, 16, 1, 1);
  packFwd(7, r1a_w, 16, 32, 3, 3);
  packFwd(8, r1b_w, 32, 16, 1, 1);
  packFwd(9, d3_w, 4, 2, 1, 1);
  {
    int sl = 10;
    for (int ph = 0; ph < 2; ++ph) for (int pw = 0; pw < 2; ++pw) packT(sl++, d0_w, 16, 8, ph, pw);
    for (int ph = 0; ph < 2; ++ph) for (int pw = 0; pw < 2; ++pw) packT(sl++, d1_w, 8, 4, ph, pw);
    for (int ph = 0; ph < 2; ++ph) for (int pw = 0; pw < 2; ++pw) packT(sl++, d2_w, 4, 4, ph, pw);
  }

  // ---- conv launch helpers ----
  auto runFwd = [&](const _Float16* in16, const float* in32, int slot,
                    const float* bias, const float* ss,
                    _Float16* out16, float* out32, const float* lref,
                    int Cin, int Cout, int IH, int OH, int KH, int pad, int stride,
                    int reluIn, int reluOut) {
    ConvP p{};
    p.in = in16; p.in32 = in32; p.wpack = pk(slot); p.bias = bias; p.ss = ss;
    p.out = out16; p.out32 = out32; p.lref = lref; p.lacc = acc + 1;
    p.N = 512; p.Cin = Cin; p.IH = IH; p.IW = IH; p.Cout = Cout;
    p.OH = OH; p.OW = OH; p.OHF = OH; p.OWF = OH;
    p.stride = stride;
    const int T = KH * KH;
    p.K = Cin * T;
    p.chunks = (p.K + 31) / 32; p.ntiles = (Cout + 15) / 16;
    p.owShift = ilog2(OH); p.ohwShift = ilog2(OH * OH);
    p.relu_in = reluIn; p.relu_out = reluOut;
    p.oh_mul = 1; p.oh_add = 0; p.ow_mul = 1; p.ow_add = 0;
    unsigned long long dyP = 0, dxP = 0;
    for (int t = 0; t < T; ++t) {
      const int fh = t / KH, fw = t - fh * KH;
      dyP |= (unsigned long long)((fh - pad + 2) & 15) << (4 * t);
      dxP |= (unsigned long long)((fw - pad + 2) & 15) << (4 * t);
    }
    p.dyP = dyP; p.dxP = dxP;
    const int tiles = (512 * OH * OH) / 16;
    const int blocks = (tiles + WAVES_PER_BLOCK - 1) / WAVES_PER_BLOCK;
    dispatch_conv(p, T, in32 != nullptr, blocks, stream);
  };
  auto runT = [&](const _Float16* in16, int slot, const float* bias, _Float16* out16,
                  int Cin, int Cout, int IH, int ph, int pw) {
    ConvP p{};
    p.in = in16; p.wpack = pk(slot); p.bias = bias; p.out = out16; p.lacc = acc + 1;
    p.N = 512; p.Cin = Cin; p.IH = IH; p.IW = IH; p.Cout = Cout;
    p.OH = IH; p.OW = IH; p.OHF = 2 * IH; p.OWF = 2 * IH;
    p.stride = 1; p.K = Cin * 4;
    p.chunks = (p.K + 31) / 32; p.ntiles = 1;
    p.owShift = ilog2(IH); p.ohwShift = ilog2(IH * IH);
    p.relu_out = 1;
    p.oh_mul = 2; p.oh_add = ph; p.ow_mul = 2; p.ow_add = pw;
    const int dH[2] = {0, ph ? 1 : -1};
    const int dW[2] = {0, pw ? 1 : -1};
    unsigned long long dyP = 0, dxP = 0;
    for (int t = 0; t < 4; ++t) {
      dyP |= (unsigned long long)((dH[t >> 1] + 2) & 15) << (4 * t);
      dxP |= (unsigned long long)((dW[t & 1] + 2) & 15) << (4 * t);
    }
    p.dyP = dyP; p.dxP = dxP;
    const int tiles = (512 * IH * IH) / 16;
    const int blocks = (tiles + WAVES_PER_BLOCK - 1) / WAVES_PER_BLOCK;
    dispatch_conv(p, 4, false, blocks, stream);
  };
  auto runStats = [&](const _Float16* act, int b, int C, int HW) {
    bn_stats<<<dim3(512 * C), 256, 0, stream>>>(act, sums(b), C - 1, HW);
  };
  auto runBNFin = [&](int b, const float* g, const float* be, int C, int HW) {
    bn_finalize<<<dim3(1), 64, 0, stream>>>(sums(b), ssb(b), g, be, C, 1.0f / (512.0f * HW));
  };

  // ---- encoder ----
  runFwd(nullptr, inputs, 0, c0_b, nullptr, z0, nullptr, nullptr, 2, 8, 128, 128, 1, 0, 1, 0, 0);
  runFwd(z0, nullptr, 1, c1_b, nullptr, y1, nullptr, nullptr, 8, 8, 128, 64, 4, 1, 2, 0, 0);
  runStats(y1, 0, 8, 4096);  runBNFin(0, bn1_g, bn1_b, 8, 4096);
  runFwd(y1, nullptr, 2, c2_b, ssb(0), y2, nullptr, nullptr, 8, 16, 64, 32, 4, 1, 2, 1, 0);
  runStats(y2, 1, 16, 1024); runBNFin(1, bn2_g, bn2_b, 16, 1024);
  runFwd(y2, nullptr, 3, c3_b, ssb(1), y3, nullptr, nullptr, 16, 16, 32, 16, 4, 1, 2, 1, 0);
  runStats(y3, 2, 16, 256);  runBNFin(2, bn3_g, bn3_b, 16, 256);
  runFwd(y3, nullptr, 4, c4_b, ssb(2), y4, nullptr, nullptr, 16, 16, 16, 16, 3, 1, 1, 1, 0);
  runStats(y4, 3, 16, 256);  runBNFin(3, bn4_g, bn4_b, 16, 256);
  bn_apply_k<<<dim3(2048), 256, 0, stream>>>(y4, ssb(3), zf, 0, 8, 15, 2097152);

  // ---- residual blocks ----
  runFwd(nullptr, zf, 5, r0a_b, nullptr, h32, nullptr, nullptr, 16, 32, 16, 16, 3, 1, 1, 1, 0);
  runStats(h32, 4, 32, 256); runBNFin(4, r0g1, r0b1, 32, 256);
  runFwd(h32, nullptr, 6, r0b_b, ssb(4), h16, nullptr, nullptr, 32, 16, 16, 16, 1, 0, 1, 1, 0);
  runStats(h16, 5, 16, 256); runBNFin(5, r0g2, r0b2, 16, 256);
  bn_apply_k<<<dim3(2048), 256, 0, stream>>>(h16, ssb(5), zf, 1, 8, 15, 2097152);

  runFwd(nullptr, zf, 7, r1a_b, nullptr, h32, nullptr, nullptr, 16, 32, 16, 16, 3, 1, 1, 1, 0);
  runStats(h32, 6, 32, 256); runBNFin(6, r1g1, r1b1, 32, 256);
  runFwd(h32, nullptr, 8, r1b_b, ssb(6), h16, nullptr, nullptr, 32, 16, 16, 16, 1, 0, 1, 1, 0);
  runStats(h16, 7, 16, 256); runBNFin(7, r1g2, r1b2, 16, 256);
  bn_apply_k<<<dim3(2048), 256, 0, stream>>>(h16, ssb(7), zf, 1, 8, 15, 2097152);

  // ---- vector quantizer ----
  vq_kernel<<<dim3(512), 256, 0, stream>>>(zf, emb, zq, acc);

  // ---- decoder (convT as 4 parity sub-convs each) ----
  {
    int sl = 10;
    for (int ph = 0; ph < 2; ++ph) for (int pw = 0; pw < 2; ++pw) runT(zq,  sl++, d0_b, yd0, 16, 8, 16, ph, pw);
    for (int ph = 0; ph < 2; ++ph) for (int pw = 0; pw < 2; ++pw) runT(yd0, sl++, d1_b, yd1, 8, 4, 32, ph, pw);
    for (int ph = 0; ph < 2; ++ph) for (int pw = 0; pw < 2; ++pw) runT(yd1, sl++, d2_b, yd2, 4, 4, 64, ph, pw);
  }
  // d3: 1x1 conv 4->2, write decoded f32 to d_out, fuse recon loss vs inputs
  runFwd(yd2, nullptr, 9, d3_b, nullptr, nullptr, dout, inputs, 4, 2, 128, 128, 1, 0, 1, 0, 0);

  final_losses<<<dim3(1), 1, 0, stream>>>(acc, dout + 16777216);
}
